// GraphAttentionNetwork_3667902071227
// MI455X (gfx1250) — matrix operand order repaired and checked by reading, not compiled, not run on hardware
//
#include <hip/hip_runtime.h>

#define DEV __device__ __forceinline__

typedef __attribute__((ext_vector_type(16))) __bf16 bf16x16;
typedef __attribute__((ext_vector_type(8)))  float  v8f;

// ---------- order-preserving float <-> uint for atomicMax-based segment max ----------
DEV unsigned f2ord(float f) {
  unsigned u = __float_as_uint(f);
  return (u & 0x80000000u) ? ~u : (u | 0x80000000u);
}
DEV float ord2f(unsigned u) {
  return __uint_as_float((u & 0x80000000u) ? (u & 0x7FFFFFFFu) : ~u);
}

// ---------------------------------- fill ----------------------------------
__global__ void fill_u32_kernel(unsigned* __restrict__ p, unsigned v, long long n) {
  long long i = (long long)blockIdx.x * blockDim.x + threadIdx.x;
  if (i < n) p[i] = v;
}

// ---- pack W[128,M] (f32 row-major) into bf16 WMMA B-fragment layout ----
// P[((kt*NT + t)*32 + lane)*16 + j] = bf16(W[(kt*32 + (j<8 ? half*8+j : 16+half*8+(j-8)))*M + t*16 + l16])
// so a wave's B fragment for (kt, t) is one contiguous 1 KB block: 32 B per lane.
template<int M>
__global__ void pack_w_kernel(const float* __restrict__ W, __bf16* __restrict__ P) {
  constexpr int NT = M / 16;
  constexpr int TOT = 4 * NT * 32 * 16;            // K/32 tiles * col tiles * lanes * elems
  const int i = blockIdx.x * blockDim.x + threadIdx.x;
  if (i >= TOT) return;
  const int j    = i & 15;
  const int lane = (i >> 4) & 31;
  const int rest = i >> 9;                          // kt_idx*NT + t
  const int t    = rest % NT;
  const int kt   = rest / NT;
  const int half = lane >> 4;
  const int l16  = lane & 15;
  const int k    = kt * 32 + (j < 8 ? half * 8 + j : 16 + half * 8 + (j - 8));
  const int col  = t * 16 + l16;
  P[i] = (__bf16)W[(size_t)k * M + col];
}

// ---- dual GEMM: Cl = A@Wl + bl, Cr = A@Wr + br; A[n,128] f32, packed bf16 weights ----
// Each wave owns a 16-row strip x (NTW*16)-column group; A fragment loaded once per
// K-tile and fed to 2*NTW WMMAs. NTW=4 keeps accumulators at 64 VGPRs -> no spills.
// A-frag (16-bit A 16x32): lane half h, elem j -> K = (j<8 ? h*8+j : 16+h*8+(j-8))
//   -> two contiguous 8-float runs -> 4x float4 loads.
// D layout (f32 16x16): VGPR r: lanes0-15 -> (row r, col l16); lanes16-31 -> (row 8+r, col l16).
template<int M, int NTW>
__global__ __launch_bounds__(256)
void gemm2_bias_wmma(const float* __restrict__ A,
                     const __bf16* __restrict__ PWl, const __bf16* __restrict__ PWr,
                     const float* __restrict__ bl, const float* __restrict__ br,
                     float* __restrict__ Cl, float* __restrict__ Cr, int nrows) {
  constexpr int K  = 128;
  constexpr int NT = M / 16;       // total col tiles
  constexpr int CG = NT / NTW;     // column groups (waves per strip)
  const int lane  = threadIdx.x & 31;
  const int wave  = blockIdx.x * (blockDim.x >> 5) + (threadIdx.x >> 5);
  const int strip = wave / CG;
  const int cg    = wave % CG;
  const int row0  = strip * 16;
  if (row0 >= nrows) return;
  const int half = lane >> 4;
  const int l16  = lane & 15;
  int arow = row0 + l16;
  if (arow >= nrows) arow = nrows - 1;   // clamp; clamped rows are never stored

  v8f zero = {};
  v8f accl[NTW], accr[NTW];
#pragma unroll
  for (int t = 0; t < NTW; ++t) { accl[t] = zero; accr[t] = zero; }

#pragma unroll
  for (int kt = 0; kt < 4; ++kt) {
    const float* ap = A + (size_t)arow * K + kt * 32 + half * 8;
    const float4 a0 = *(const float4*)(ap + 0);
    const float4 a1 = *(const float4*)(ap + 4);
    const float4 a2 = *(const float4*)(ap + 16);
    const float4 a3 = *(const float4*)(ap + 20);
    bf16x16 afrag;
    afrag[0]  = (__bf16)a0.x; afrag[1]  = (__bf16)a0.y;
    afrag[2]  = (__bf16)a0.z; afrag[3]  = (__bf16)a0.w;
    afrag[4]  = (__bf16)a1.x; afrag[5]  = (__bf16)a1.y;
    afrag[6]  = (__bf16)a1.z; afrag[7]  = (__bf16)a1.w;
    afrag[8]  = (__bf16)a2.x; afrag[9]  = (__bf16)a2.y;
    afrag[10] = (__bf16)a2.z; afrag[11] = (__bf16)a2.w;
    afrag[12] = (__bf16)a3.x; afrag[13] = (__bf16)a3.y;
    afrag[14] = (__bf16)a3.z; afrag[15] = (__bf16)a3.w;

#pragma unroll
    for (int tt = 0; tt < NTW; ++tt) {
      const int t = cg * NTW + tt;
      const size_t foff = (((size_t)kt * NT + t) * 32 + lane) * 16;
      const bf16x16 bfl = *(const bf16x16*)(PWl + foff);
      accl[tt] = __builtin_amdgcn_wmma_f32_16x16x32_bf16(
          false, afrag, false, bfl, (short)0, accl[tt], false, false);
      const bf16x16 bfr = *(const bf16x16*)(PWr + foff);
      accr[tt] = __builtin_amdgcn_wmma_f32_16x16x32_bf16(
          false, afrag, false, bfr, (short)0, accr[tt], false, false);
    }
  }

#pragma unroll
  for (int tt = 0; tt < NTW; ++tt) {
    const int col = (cg * NTW + tt) * 16 + l16;
    const float vbl = bl[col];
    const float vbr = br[col];
#pragma unroll
    for (int r = 0; r < 8; ++r) {
      const int row = row0 + half * 8 + r;
      if (row < nrows) {
        Cl[(size_t)row * M + col] = accl[tt][r] + vbl;
        Cr[(size_t)row * M + col] = accr[tt][r] + vbr;
      }
    }
  }
}

// ---------- pass A: per-(edge,head) score + segment max (atomicMax) ----------
// TPE threads cooperate on one (e,h): each handles C/TPE contiguous channels, shfl-reduce.
template<int H, int C, int TPE>
__global__ void edge_score_kernel(const float* __restrict__ xl, const float* __restrict__ xr,
                                  const int* __restrict__ ei, const float* __restrict__ att,
                                  float* __restrict__ score, unsigned* __restrict__ smax, int nE) {
  constexpr int CS = C / TPE;
  const long long gid = (long long)blockIdx.x * blockDim.x + threadIdx.x;
  const long long idx = gid / TPE;
  const int sub = (int)(gid % TPE);
  if (idx >= (long long)nE * H) return;
  const int e = (int)(idx / H), h = (int)(idx % H);
  const int src = ei[e];
  const int dst = ei[nE + e];
  const float* pl = xl + (size_t)src * (H * C) + h * C + sub * CS;
  const float* pr = xr + (size_t)dst * (H * C) + h * C + sub * CS;
  const float* pa = att + h * C + sub * CS;
  float s = 0.f;
#pragma unroll
  for (int c = 0; c < CS; c += 4) {
    float4 a = *(const float4*)(pl + c);
    float4 b = *(const float4*)(pr + c);
    float4 w = *(const float4*)(pa + c);
    float v;
    v = a.x + b.x; v = v > 0.f ? v : 0.2f * v; s += v * w.x;
    v = a.y + b.y; v = v > 0.f ? v : 0.2f * v; s += v * w.y;
    v = a.z + b.z; v = v > 0.f ? v : 0.2f * v; s += v * w.z;
    v = a.w + b.w; v = v > 0.f ? v : 0.2f * v; s += v * w.w;
  }
#pragma unroll
  for (int off = 1; off < TPE; off <<= 1) s += __shfl_xor(s, off, 32);
  if (sub == 0) {
    score[idx] = s;
    atomicMax(&smax[(size_t)dst * H + h], f2ord(s));
  }
}

// ---------- pass B: exp(score - max) + segment sum of denominators ----------
template<int H>
__global__ void edge_exp_kernel(const int* __restrict__ ei, float* __restrict__ score,
                                const unsigned* __restrict__ smax, float* __restrict__ denom,
                                int nE) {
  const long long idx = (long long)blockIdx.x * blockDim.x + threadIdx.x;
  if (idx >= (long long)nE * H) return;
  const int e = (int)(idx / H), h = (int)(idx % H);
  const int dst = ei[nE + e];
  const float m  = ord2f(smax[(size_t)dst * H + h]);
  const float ex = __expf(score[idx] - m);
  score[idx] = ex;
  atomicAdd(&denom[(size_t)dst * H + h], ex);
}

// ---------- pass C: scatter alpha * xl[src] into out[dst] ----------
template<int H, int C, int TPE>
__global__ void edge_scatter_kernel(const float* __restrict__ xl, const int* __restrict__ ei,
                                    const float* __restrict__ score, const float* __restrict__ denom,
                                    float* __restrict__ out, int nE) {
  constexpr int CS = C / TPE;
  const long long gid = (long long)blockIdx.x * blockDim.x + threadIdx.x;
  const long long idx = gid / TPE;
  const int sub = (int)(gid % TPE);
  if (idx >= (long long)nE * H) return;
  const int e = (int)(idx / H), h = (int)(idx % H);
  const int src = ei[e];
  const int dst = ei[nE + e];
  const float alpha = score[idx] / denom[(size_t)dst * H + h];
  const float* pl = xl + (size_t)src * (H * C) + h * C + sub * CS;
  float* po = out + (size_t)dst * (H * C) + h * C + sub * CS;
#pragma unroll
  for (int c = 0; c < CS; ++c) atomicAdd(&po[c], alpha * pl[c]);
}

// ---------- h = elu(acc + bias) (in place) ----------
template<int M>
__global__ void bias_elu_kernel(float* __restrict__ h, const float* __restrict__ bias, long long n) {
  const long long i = (long long)blockIdx.x * blockDim.x + threadIdx.x;
  if (i >= n) return;
  const float v = h[i] + bias[(int)(i & (M - 1))];
  h[i] = v > 0.f ? v : (__expf(v) - 1.f);
}

// ---------- out += bias (final) ----------
template<int M>
__global__ void bias_add_kernel(float* __restrict__ o, const float* __restrict__ bias, long long n) {
  const long long i = (long long)blockIdx.x * blockDim.x + threadIdx.x;
  if (i >= n) return;
  o[i] += bias[(int)(i & (M - 1))];
}

static inline int cdiv_i(long long a, long long b) { return (int)((a + b - 1) / b); }

extern "C" void kernel_launch(void* const* d_in, const int* in_sizes, int n_in,
                              void* d_out, int out_size, void* d_ws, size_t ws_size,
                              hipStream_t stream) {
  const float* x     = (const float*)d_in[0];
  const int*   ei    = (const int*)d_in[1];     // edge_index [2,E] (int32 on device)
  const float* Wl1   = (const float*)d_in[2];
  const float* bl1   = (const float*)d_in[3];
  const float* Wr1   = (const float*)d_in[4];
  const float* br1   = (const float*)d_in[5];
  const float* att1  = (const float*)d_in[6];
  const float* bias1 = (const float*)d_in[7];
  const float* Wl2   = (const float*)d_in[8];
  const float* bl2   = (const float*)d_in[9];
  const float* Wr2   = (const float*)d_in[10];
  const float* br2   = (const float*)d_in[11];
  const float* att2  = (const float*)d_in[12];
  const float* bias2 = (const float*)d_in[13];

  const int N = in_sizes[0] / 128;   // 100000
  const int E = in_sizes[1] / 2;     // 1000000

  // -------- workspace layout --------
  char* ws = (char*)d_ws;
  const size_t szX = (size_t)N * 128 * sizeof(float);        // 51.2 MB
  float*    XL    = (float*)(ws);                             // xl (layer1: N*128, layer2: N*64)
  float*    XR    = (float*)(ws + szX);                       // xr
  float*    ACC   = (float*)(ws + 2 * szX);                   // layer1 accum -> h (elu)
  float*    SCORE = (float*)(ws + 3 * szX);                   // E*8 (layer1) / E (layer2)
  unsigned* SMAX  = (unsigned*)(ws + 3 * szX + (size_t)E * 8 * sizeof(float));
  float*    DENOM = (float*)((char*)SMAX + (size_t)N * 8 * sizeof(float));
  __bf16*   PWl1  = (__bf16*)((char*)DENOM + (size_t)N * 8 * sizeof(float));
  __bf16*   PWr1  = PWl1 + 128 * 128;
  __bf16*   PWl2  = PWr1 + 128 * 128;
  __bf16*   PWr2  = PWl2 + 128 * 64;
  float*    OUT   = (float*)d_out;

  const int B = 256;
  const int strips = (N + 15) / 16;
  const int gemm_blocks1 = cdiv_i((long long)strips * 2, 8);  // M=128: 2 col groups per strip
  const int gemm_blocks2 = cdiv_i(strips, 8);                 // M=64 : 1 col group per strip

  // -------- pack all weights into bf16 WMMA B-fragment layout (tiny) --------
  pack_w_kernel<128><<<cdiv_i(128 * 128, B), B, 0, stream>>>(Wl1, PWl1);
  pack_w_kernel<128><<<cdiv_i(128 * 128, B), B, 0, stream>>>(Wr1, PWr1);
  pack_w_kernel<64><<<cdiv_i(128 * 64, B), B, 0, stream>>>(Wl2, PWl2);
  pack_w_kernel<64><<<cdiv_i(128 * 64, B), B, 0, stream>>>(Wr2, PWr2);

  // ================= layer 1: 128 -> 8 heads x 16 =================
  gemm2_bias_wmma<128, 4><<<gemm_blocks1, B, 0, stream>>>(x, PWl1, PWr1, bl1, br1, XL, XR, N);

  fill_u32_kernel<<<cdiv_i((long long)N * 8, B), B, 0, stream>>>(SMAX, 0u, (long long)N * 8);
  fill_u32_kernel<<<cdiv_i((long long)N * 8, B), B, 0, stream>>>((unsigned*)DENOM, 0u, (long long)N * 8);
  fill_u32_kernel<<<cdiv_i((long long)N * 128, B), B, 0, stream>>>((unsigned*)ACC, 0u, (long long)N * 128);

  edge_score_kernel<8, 16, 1><<<cdiv_i((long long)E * 8, B), B, 0, stream>>>(XL, XR, ei, att1, SCORE, SMAX, E);
  edge_exp_kernel<8><<<cdiv_i((long long)E * 8, B), B, 0, stream>>>(ei, SCORE, SMAX, DENOM, E);
  edge_scatter_kernel<8, 16, 1><<<cdiv_i((long long)E * 8, B), B, 0, stream>>>(XL, ei, SCORE, DENOM, ACC, E);

  bias_elu_kernel<128><<<cdiv_i((long long)N * 128, B), B, 0, stream>>>(ACC, bias1, (long long)N * 128);

  // ================= layer 2: 128 -> 1 head x 64 =================
  gemm2_bias_wmma<64, 4><<<gemm_blocks2, B, 0, stream>>>(ACC, PWl2, PWr2, bl2, br2, XL, XR, N);

  fill_u32_kernel<<<cdiv_i(N, B), B, 0, stream>>>(SMAX, 0u, N);
  fill_u32_kernel<<<cdiv_i(N, B), B, 0, stream>>>((unsigned*)DENOM, 0u, N);
  fill_u32_kernel<<<cdiv_i((long long)N * 64, B), B, 0, stream>>>((unsigned*)OUT, 0u, (long long)N * 64);

  edge_score_kernel<1, 64, 4><<<cdiv_i((long long)E * 4, B), B, 0, stream>>>(XL, XR, ei, att2, SCORE, SMAX, E);
  edge_exp_kernel<1><<<cdiv_i((long long)E, B), B, 0, stream>>>(ei, SCORE, SMAX, DENOM, E);
  edge_scatter_kernel<1, 64, 4><<<cdiv_i((long long)E * 4, B), B, 0, stream>>>(XL, ei, SCORE, DENOM, OUT, E);

  bias_add_kernel<64><<<cdiv_i((long long)N * 64, B), B, 0, stream>>>(OUT, bias2, (long long)N * 64);
}
